// SieveKAN_39633958207771
// MI455X (gfx1250) — matrix operand order repaired and checked by reading, not compiled, not run on hardware
//
#include <hip/hip_runtime.h>

// ---------------------------------------------------------------------------
// KAN forward for MI455X (gfx1250, wave32).
//  - Spline path: structurally-sparse (4 taps) gather-FMA from an L2-resident
//    transposed weight layout swT[i][t][o]; tap-row offsets are wave-uniform
//    and forced into SGPRs via readfirstlane so the hot loop is
//    4x global_load_b128 (saddr form) + 16x v_fmac per 512 MACs.
//  - Base path: dense SiLU(X) @ bw.T via v_wmma_f32_16x16x32_f16, with bw
//    pre-swizzled into fragment-ready f16 so B-fragments are contiguous loads.
// ---------------------------------------------------------------------------

typedef __attribute__((ext_vector_type(16))) _Float16 v16h;
typedef __attribute__((ext_vector_type(8)))  _Float16 v8h;
typedef __attribute__((ext_vector_type(8)))  float    v8f;

union FragU { v16h v; v8h h[2]; };

// Problem constants
#define BATCH 2048
#define WID   256
#define NBAS  254   // G + order
#define BT    16    // batch rows per block

// Workspace layout (bytes)
#define SWT_BYTES (254ull * 256ull * 256ull * 4ull)          // 66,584,576
#define BWF_BYTES (8ull * 16ull * 32ull * 16ull * 2ull)      // 131,072
#define HA_OFF    (SWT_BYTES + BWF_BYTES)
#define HB_OFF    (HA_OFF + (size_t)BATCH * WID * 4ull)

__device__ __forceinline__ float silu_f(float x) {
    return x / (1.0f + __expf(-x));
}

__device__ __forceinline__ float pick4(int r, float w0, float w1, float w2, float w3) {
    float t = (r == 0) ? w0 : 0.0f;
    t = (r == 1) ? w1 : t;
    t = (r == 2) ? w2 : t;
    t = (r == 3) ? w3 : t;
    return t;
}

// Closed-form uniform cubic B-spline: 4 weights for columns jb..jb+3 of the
// 254-wide basis, matching the reference's Cox-de Boor on the uniform grid
// t_j = -0.05 + (j-3)*H, H = 1.1/251. Out-of-range x -> all-zero weights.
__device__ __forceinline__ void basis4(float x, float4& v, int& jb) {
    const float INVH = 251.0f / 1.1f;
    float u  = (x + 0.05f) * INVH + 3.0f;
    float mf = floorf(u);
    float f  = u - mf;
    bool ok  = (u >= 0.0f) && (u < 257.0f);   // interval m in [0,256]; NaN -> false
    float f2 = f * f, f3 = f2 * f;
    float om = 1.0f - f;
    float w0 = om * om * om * (1.0f / 6.0f);
    float w1 = (3.0f * f3 - 6.0f * f2 + 4.0f) * (1.0f / 6.0f);
    float w2 = (-3.0f * f3 + 3.0f * f2 + 3.0f * f + 1.0f) * (1.0f / 6.0f);
    float w3 = f3 * (1.0f / 6.0f);
    int m  = (int)mf;
    int j0 = m - 3;
    int jc = j0 < 0 ? 0 : (j0 > 250 ? 250 : j0);
    int d  = j0 - jc;                          // shift for edge clipping
    v.x = ok ? pick4(0 - d, w0, w1, w2, w3) : 0.0f;
    v.y = ok ? pick4(1 - d, w0, w1, w2, w3) : 0.0f;
    v.z = ok ? pick4(2 - d, w0, w1, w2, w3) : 0.0f;
    v.w = ok ? pick4(3 - d, w0, w1, w2, w3) : 0.0f;
    jb = jc;
}

// swT[i][t][o] <- sw[o][i][t]  (makes the gather coalesced over outputs)
__global__ __launch_bounds__(256) void transpose_sw(const float* __restrict__ sw,
                                                    float* __restrict__ swT, int IN) {
    int i  = blockIdx.x >> 2;
    int t0 = (blockIdx.x & 3) * 64;
    int t1 = (t0 + 64 > NBAS) ? NBAS : (t0 + 64);
    int o  = threadIdx.x;
    const float* src = sw + ((size_t)o * IN + i) * NBAS;
    for (int t = t0; t < t1; ++t)
        swT[(((size_t)i * NBAS + t) << 8) + o] = src[t];
}

// bwF: fragment-ready f16 B-matrix tiles. Layout [kc][ntile][lane][16 halves]:
// lane holds N = ntile*16 + (lane&15), K = (lane>=16 ? 16:0)+e  (e = 0..15),
// contiguous k order, same as the A loads in kan_layer. B[k][n] = bw[n][k] (bw.T).
__global__ __launch_bounds__(256) void make_bwF(const float* __restrict__ bw,
                                                _Float16* __restrict__ bwF, int IN) {
    int idx = blockIdx.x * 256 + threadIdx.x;
    int e   = idx & 15;
    int L   = (idx >> 4) & 31;
    int nt  = (idx >> 9) & 15;
    int kc  = idx >> 13;
    int k   = kc * 32 + ((L >> 4) << 4) + e;
    int n   = nt * 16 + (L & 15);
    bwF[idx] = (_Float16)bw[(size_t)n * IN + k];
}

// Fused KAN layer: out[b,o] = wb*base + ws*spline (+ x residual). OUT = 256.
template <int IN, bool RES>
__global__ __launch_bounds__(256) void kan_layer(const float* __restrict__ X,
                                                 const float* __restrict__ swT,
                                                 const _Float16* __restrict__ bwF,
                                                 const float* __restrict__ wbp,
                                                 const float* __restrict__ wsp,
                                                 float* __restrict__ Y) {
    constexpr int KC = IN / 32;
    __shared__ __attribute__((aligned(16))) _Float16 Abuf[BT][IN + 8]; // +8: bank spread
    __shared__ __attribute__((aligned(16))) float    Bse[BT][256];     // WMMA base result
    __shared__ __attribute__((aligned(16))) float4   Wbuf[BT][64];     // basis weights
    __shared__ int                                   Jbuf[BT][64];     // tap-row byte/4 offset

    const int tid  = threadIdx.x;
    const int lane = tid & 31;
    const int wv   = tid >> 5;            // 8 wave32s
    const int b0   = blockIdx.x << 4;     // 16 batch rows / block

    // ---- Phase 1: SiLU(x) -> f16 A-buffer in LDS --------------------------
    for (int idx = tid; idx < BT * IN; idx += 256) {
        int b = idx / IN, i = idx % IN;
        float x = X[(size_t)(b0 + b) * IN + i];
        Abuf[b][i] = (_Float16)silu_f(x);
    }
    __syncthreads();

    // ---- Phase 2: base path via WMMA (2 N-tiles per wave, 1 M-tile) -------
    {
        const int hs   = lane >> 4;       // lane-half select
        const int mrow = lane & 15;       // A-matrix M = lane&15
        v8f cacc0 = {};
        v8f cacc1 = {};
#pragma unroll
        for (int kc = 0; kc < KC; ++kc) {
            int colbase = kc * 32 + hs * 16;
            FragU a;
            a.h[0] = *(const v8h*)&Abuf[mrow][colbase];        // K {0..7}/{16..23}
            a.h[1] = *(const v8h*)&Abuf[mrow][colbase + 8];    // +8
            const v16h bf0 = *(const v16h*)(bwF + (size_t)(((kc * 16 + (wv * 2 + 0)) * 32 + lane) << 4));
            const v16h bf1 = *(const v16h*)(bwF + (size_t)(((kc * 16 + (wv * 2 + 1)) * 32 + lane) << 4));
            cacc0 = __builtin_amdgcn_wmma_f32_16x16x32_f16(false, a.v, false, bf0,
                                                           (short)0, cacc0, false, false);
            cacc1 = __builtin_amdgcn_wmma_f32_16x16x32_f16(false, a.v, false, bf1,
                                                           (short)0, cacc1, false, false);
        }
        // C/D layout: VGPR r -> M=r (lanes 0-15) / M=8+r (lanes 16-31), N=lane&15
#pragma unroll
        for (int r = 0; r < 8; ++r) {
            int m = r + hs * 8;
            Bse[m][(wv * 2 + 0) * 16 + (lane & 15)] = cacc0[r];
            Bse[m][(wv * 2 + 1) * 16 + (lane & 15)] = cacc1[r];
        }
    }

    // ---- Phase 3: spline gather (4-tap sparse) ----------------------------
    const int ohalf = wv & 1;
    const int bq    = wv >> 1;
    const int obase = ohalf * 128 + lane * 4;  // 4 consecutive outputs per lane
    float4 acc4[4];
#pragma unroll
    for (int bi = 0; bi < 4; ++bi) acc4[bi] = make_float4(0.f, 0.f, 0.f, 0.f);

    for (int i0 = 0; i0 < IN; i0 += 64) {
        __syncthreads();   // previous chunk's gather done before overwrite
        for (int p = tid; p < BT * 64; p += 256) {
            int b = p >> 6, ic = p & 63;
            float x = X[(size_t)(b0 + b) * IN + i0 + ic];
            float4 v; int jb;
            basis4(x, v, jb);
            Wbuf[b][ic] = v;
            // full tap-row offset (in floats): rows jb..jb+3 of feature i0+ic
            Jbuf[b][ic] = ((i0 + ic) * NBAS + jb) << 8;
        }
        __syncthreads();

#pragma unroll
        for (int bi = 0; bi < 4; ++bi) {
            int b = bq * 4 + bi;
#pragma unroll 4
            for (int ic = 0; ic < 64; ++ic) {
                // wave-uniform (all lanes read the same LDS word) -> SGPR base
                unsigned off = (unsigned)__builtin_amdgcn_readfirstlane(Jbuf[b][ic]);
                float4 w  = Wbuf[b][ic];
                const float* pu = swT + off;        // uniform row base in SGPRs
                float4 r0 = *(const float4*)(pu + obase);
                float4 r1 = *(const float4*)(pu + obase + 256);
                float4 r2 = *(const float4*)(pu + obase + 512);
                float4 r3 = *(const float4*)(pu + obase + 768);
                float4 a = acc4[bi];
                a.x = fmaf(w.x, r0.x, a.x); a.y = fmaf(w.x, r0.y, a.y);
                a.z = fmaf(w.x, r0.z, a.z); a.w = fmaf(w.x, r0.w, a.w);
                a.x = fmaf(w.y, r1.x, a.x); a.y = fmaf(w.y, r1.y, a.y);
                a.z = fmaf(w.y, r1.z, a.z); a.w = fmaf(w.y, r1.w, a.w);
                a.x = fmaf(w.z, r2.x, a.x); a.y = fmaf(w.z, r2.y, a.y);
                a.z = fmaf(w.z, r2.z, a.z); a.w = fmaf(w.z, r2.w, a.w);
                a.x = fmaf(w.w, r3.x, a.x); a.y = fmaf(w.w, r3.y, a.y);
                a.z = fmaf(w.w, r3.z, a.z); a.w = fmaf(w.w, r3.w, a.w);
                acc4[bi] = a;
            }
        }
    }

    // ---- Phase 4: combine base + spline (+ residual), write out -----------
    const float wbv = wbp[0];
    const float wsv = wsp[0];
#pragma unroll
    for (int bi = 0; bi < 4; ++bi) {
        int b = bq * 4 + bi;
        float4 bse = *(const float4*)&Bse[b][obase];
        float4 r;
        r.x = fmaf(wbv, bse.x, wsv * acc4[bi].x);
        r.y = fmaf(wbv, bse.y, wsv * acc4[bi].y);
        r.z = fmaf(wbv, bse.z, wsv * acc4[bi].z);
        r.w = fmaf(wbv, bse.w, wsv * acc4[bi].w);
        if constexpr (RES) {
            float4 xr = *(const float4*)(X + (size_t)(b0 + b) * IN + obase);
            r.x += xr.x; r.y += xr.y; r.z += xr.z; r.w += xr.w;
        }
        *(float4*)(Y + (size_t)(b0 + b) * 256 + obase) = r;
    }
}

// Output layer: 256 -> 1. One wave32 per batch row, lane-parallel over inputs.
__global__ __launch_bounds__(256) void kan_out(const float* __restrict__ H,
                                               const float* __restrict__ bw,
                                               const float* __restrict__ sw,
                                               const float* __restrict__ wbp,
                                               const float* __restrict__ wsp,
                                               float* __restrict__ out) {
    int lane = threadIdx.x & 31;
    int b    = blockIdx.x * 8 + (threadIdx.x >> 5);
    float accB = 0.0f, accS = 0.0f;
#pragma unroll
    for (int ii = 0; ii < 8; ++ii) {
        int i = ii * 32 + lane;
        float x = H[(size_t)b * 256 + i];
        accB = fmaf(silu_f(x), bw[i], accB);
        float4 v; int jb;
        basis4(x, v, jb);
        const float* p = sw + (size_t)i * NBAS + jb;
        accS = fmaf(v.x, p[0], accS);
        accS = fmaf(v.y, p[1], accS);
        accS = fmaf(v.z, p[2], accS);
        accS = fmaf(v.w, p[3], accS);
    }
#pragma unroll
    for (int off = 16; off > 0; off >>= 1) {
        accB += __shfl_down(accB, off);
        accS += __shfl_down(accS, off);
    }
    if (lane == 0) out[b] = wbp[0] * accB + wsp[0] * accS;
}

extern "C" void kernel_launch(void* const* d_in, const int* in_sizes, int n_in,
                              void* d_out, int out_size, void* d_ws, size_t ws_size,
                              hipStream_t stream) {
    (void)in_sizes; (void)n_in; (void)out_size; (void)ws_size;
    const float* x   = (const float*)d_in[0];
    const float* bw0 = (const float*)d_in[1];
    const float* sw0 = (const float*)d_in[2];
    const float* wb0 = (const float*)d_in[3];
    const float* ws0 = (const float*)d_in[4];
    const float* bw1 = (const float*)d_in[5];
    const float* sw1 = (const float*)d_in[6];
    const float* wb1 = (const float*)d_in[7];
    const float* ws1 = (const float*)d_in[8];
    const float* bw2 = (const float*)d_in[9];
    const float* sw2 = (const float*)d_in[10];
    const float* wb2 = (const float*)d_in[11];
    const float* ws2 = (const float*)d_in[12];
    const float* bw3 = (const float*)d_in[13];
    const float* sw3 = (const float*)d_in[14];
    const float* wb3 = (const float*)d_in[15];
    const float* ws3 = (const float*)d_in[16];

    char*      ws  = (char*)d_ws;
    float*     swT = (float*)ws;
    _Float16*  bwF = (_Float16*)(ws + SWT_BYTES);
    float*     hA  = (float*)(ws + HA_OFF);
    float*     hB  = (float*)(ws + HB_OFF);

    // Layer 0: 64 -> 256
    transpose_sw<<<64 * 4, 256, 0, stream>>>(sw0, swT, 64);
    make_bwF<<<64, 256, 0, stream>>>(bw0, bwF, 64);
    kan_layer<64, false><<<BATCH / BT, 256, 0, stream>>>(x, swT, bwF, wb0, ws0, hA);

    // Layer 1: 256 -> 256, residual
    transpose_sw<<<256 * 4, 256, 0, stream>>>(sw1, swT, 256);
    make_bwF<<<256, 256, 0, stream>>>(bw1, bwF, 256);
    kan_layer<256, true><<<BATCH / BT, 256, 0, stream>>>(hA, swT, bwF, wb1, ws1, hB);

    // Layer 2: 256 -> 256, residual
    transpose_sw<<<256 * 4, 256, 0, stream>>>(sw2, swT, 256);
    make_bwF<<<256, 256, 0, stream>>>(bw2, bwF, 256);
    kan_layer<256, true><<<BATCH / BT, 256, 0, stream>>>(hB, swT, bwF, wb2, ws2, hA);

    // Layer 3: 256 -> 1
    kan_out<<<BATCH / 8, 256, 0, stream>>>(hA, bw3, sw3, wb3, ws3, (float*)d_out);
}
